// RectMZI_83657372991755
// MI455X (gfx1250) — compile-verified, matrix operand order checked by reading, not run on hardware
//
#include <hip/hip_runtime.h>
#include <hip/hip_bf16.h>

#define N 256
#define NUM_UNITS (N * (N - 1) / 2)   // 32640
#define BATCH 131072
#define USE_TDM 1

typedef __attribute__((ext_vector_type(16))) __bf16 v16bf;
typedef __attribute__((ext_vector_type(8)))  float  v8f;
typedef __attribute__((ext_vector_type(4)))  unsigned int u32x4;
typedef __attribute__((ext_vector_type(8)))  unsigned int u32x8;

// ---------------- helpers ----------------

__device__ __forceinline__ unsigned int pk_bf16(float a, float b) {
#if __has_builtin(__builtin_amdgcn_cvt_pk_bf16_f32)
    auto p = __builtin_amdgcn_cvt_pk_bf16_f32(a, b);
    unsigned int r;
    __builtin_memcpy(&r, &p, 4);
    return r;
#else
    // round-to-nearest-even f32 -> bf16, packed pair (a low 16, b high 16)
    unsigned int ua = __float_as_uint(a);
    unsigned int ub = __float_as_uint(b);
    ua += 0x7FFFu + ((ua >> 16) & 1u);
    ub += 0x7FFFu + ((ub >> 16) & 1u);
    return (ua >> 16) | (ub & 0xFFFF0000u);
#endif
}

__device__ __forceinline__ unsigned short f2bf(float a) {
    unsigned int ua = __float_as_uint(a);
    ua += 0x7FFFu + ((ua >> 16) & 1u);
    return (unsigned short)(ua >> 16);
}

// ---------------- kernel 1: per-unit 2x2 complex MZI blocks ----------------
__global__ __launch_bounds__(256) void mzi_kernel(
    const float* __restrict__ PHIs, const float* __restrict__ THETAs,
    const float* __restrict__ DCs, float* __restrict__ mzi) {
    int u = blockIdx.x * 256 + threadIdx.x;
    if (u >= NUM_UNITS) return;
    float phi = PHIs[u], th = THETAs[u];
    float t0 = DCs[2 * u], t1 = DCs[2 * u + 1];
    float k0 = sqrtf(fmaxf(0.0f, 1.0f - t0 * t0));
    float k1 = sqrtf(fmaxf(0.0f, 1.0f - t1 * t1));
    float sphi, cphi, sth, cth;
    __sincosf(phi, &sphi, &cphi);
    __sincosf(th, &sth, &cth);
    float A = t1 * t0, B = k1 * k0, C = t1 * k0, D = k1 * t0;
    float xr = A * cth - B, xi = A * sth;                       // m00 = e^{i phi}(A e^{i th} - B)
    float m00r = cphi * xr - sphi * xi, m00i = cphi * xi + sphi * xr;
    float yr = C * cth + D, yi = C * sth;                       // m01 = i(C e^{i th} + D)
    float m01r = -yi, m01i = yr;
    float zr = D * cth + C, zi = D * sth;                       // m10 = i e^{i phi}(D e^{i th} + C)
    float wr = cphi * zr - sphi * zi, wi = cphi * zi + sphi * zr;
    float m10r = -wi, m10i = wr;
    float m11r = A - B * cth, m11i = -B * sth;                  // m11 = A - B e^{i th}
    float* o = mzi + (size_t)u * 8;
    o[0] = m00r; o[1] = m00i; o[2] = m01r; o[3] = m01i;
    o[4] = m10r; o[5] = m10i; o[6] = m11r; o[7] = m11i;
}

// ---------------- kernel 2: propagate identity columns through 256 stages ----------------
#define COLS 8
__global__ __launch_bounds__(256) void phasor_kernel(
    const float* __restrict__ mzi, unsigned short* __restrict__ powerBf) {
    __shared__ float sRe[N * COLS];
    __shared__ float sIm[N * COLS];
    int tid = threadIdx.x;
    int j0 = blockIdx.x * COLS;
    for (int w = tid; w < N * COLS; w += 256) {
        int row = w >> 3, c = w & 7;
        sRe[w] = (row == j0 + c) ? 1.0f : 0.0f;
        sIm[w] = 0.0f;
    }
    for (int s = 0; s < N; ++s) {
        __syncthreads();
        int p = s & 1;
        int np = 128 - p;
        int idx0 = 128 * ((s + 1) >> 1) + 127 * (s >> 1);
        for (int w = tid; w < np * COLS; w += 256) {
            int pair = w >> 3, c = w & 7;
            int wg = 2 * pair + p;
            const float* m = mzi + (size_t)(idx0 + pair) * 8;
            float m00r = m[0], m00i = m[1], m01r = m[2], m01i = m[3];
            float m10r = m[4], m10i = m[5], m11r = m[6], m11i = m[7];
            int ia = wg * COLS + c, ib = ia + COLS;
            float ar = sRe[ia], ai = sIm[ia];
            float br = sRe[ib], bi = sIm[ib];
            sRe[ia] = m00r * ar - m00i * ai + m01r * br - m01i * bi;
            sIm[ia] = m00r * ai + m00i * ar + m01r * bi + m01i * br;
            sRe[ib] = m10r * ar - m10i * ai + m11r * br - m11i * bi;
            sIm[ib] = m10r * ai + m10i * ar + m11r * bi + m11i * br;
        }
    }
    __syncthreads();
    for (int w = tid; w < N * COLS; w += 256) {
        int row = w >> 3, c = w & 7;
        float re = sRe[w], im = sIm[w];
        powerBf[row * N + j0 + c] = f2bf(re * re + im * im);   // power[i][j]
    }
}

// ---------------- kernel 3: out = in(131072x256) * power^T via bf16 WMMA ----------------
// Block: 256 thr = 8 waves; block tile 128 rows x 256 cols; per-wave 16 rows x 16 N-tiles.
// B slab (256 n x 32 k bf16, LDS pitch 40 elem) staged by the Tensor Data Mover,
// double-buffered: TDM copies slab i+1 while waves run WMMAs on slab i.
#define BPITCH 40
#define SLAB_USHORT (N * BPITCH)      // 10240 elems = 20 KB

#if USE_TDM
// Issue a TDM 2D tile load: 32 bf16 wide x 256 rows from power[.][k0..k0+32)
// into LDS at ldsAddr, HW-padded 16B after every 64B row -> 80B (40-elem) pitch.
__device__ __forceinline__ void tdm_load_slab(const unsigned short* powerBf, int k0,
                                              unsigned int ldsAddr) {
    unsigned long long ga = (unsigned long long)(uintptr_t)(powerBf + k0); // tile start
    u32x4 g0 = {
        1u,                                        // [1:0] count=1 (valid user D#)
        ldsAddr,                                   // [63:32] lds_addr (bytes)
        (unsigned int)ga,                          // [95:64] global_addr[31:0]
        (unsigned int)(ga >> 32) | (2u << 30)      // [120:96] ga[56:32] | type=2
    };
    u32x8 g1 = {
        (1u << 16)                                 // data_size: 1 -> 2 bytes
            | (1u << 20)                           // pad_enable
            | (3u << 22)                           // pad_interval: 3 -> every 16 DWORDs (64B)
            | (3u << 25),                          // pad_amount:   3 -> 4 DWORDs (16B)
        (256u & 0xFFFFu) << 16,                    // tensor_dim0[15:0] = 256 (row length)
        (256u & 0xFFFFu) << 16,                    // tensor_dim1[15:0] = 256 (rows)
        32u << 16,                                 // tile_dim0 = 32 elems (64B row chunk)
        256u,                                      // tile_dim1 = 256 rows
        256u,                                      // tensor_dim0_stride[31:0] = 256 elems
        0u,                                        // dim0_stride[47:32]=0, dim1_stride lo=0
        0u
    };
    asm volatile("tensor_load_to_lds %0, %1" :: "s"(g0), "s"(g1) : "memory");
}
#endif

__global__ __launch_bounds__(256) void gemm_kernel(
    const float* __restrict__ in, const unsigned short* __restrict__ powerBf,
    float* __restrict__ out) {
    __shared__ __align__(16) unsigned short sB[2][SLAB_USHORT];   // 40 KB double buffer

    const int tid   = threadIdx.x;
    const int wave  = tid >> 5;
    const int lane  = tid & 31;
    const int r     = lane & 15;      // M (A) / N (B) within tile
    const int khSel = lane >> 4;      // 0: K 0-7,16-23 ; 1: K 8-15,24-31
    const int rowBase = blockIdx.x * 128 + wave * 16;
    const float* aRow = in + (size_t)(rowBase + r) * N;

    v8f zero = {0.f, 0.f, 0.f, 0.f, 0.f, 0.f, 0.f, 0.f};
    v8f acc[16];
#pragma unroll
    for (int t = 0; t < 16; ++t) acc[t] = zero;

#if USE_TDM
    const unsigned int ldsBase = (unsigned int)(uintptr_t)&sB[0][0];
    if (wave == 0) tdm_load_slab(powerBf, 0, ldsBase);            // prefetch slab 0
#endif

    for (int i = 0; i < 8; ++i) {
        const int k0 = i * 32;
#if USE_TDM
        if (wave == 0) __builtin_amdgcn_s_wait_tensorcnt(0);      // slab i landed
        __syncthreads();                                          // publish; prev compute done
        if (wave == 0 && i < 7)                                   // overlap: copy slab i+1
            tdm_load_slab(powerBf, k0 + 32,
                          ldsBase + (unsigned int)(((i + 1) & 1) * SLAB_USHORT * 2));
        const unsigned short* sbuf = &sB[i & 1][0];
#else
        __syncthreads();
#pragma unroll
        for (int p = 0; p < 4; ++p) {
            int c = tid + p * 256;
            int row = c >> 2, seg = c & 3;
            const uint4 g = *reinterpret_cast<const uint4*>(powerBf + row * N + k0 + seg * 8);
            *reinterpret_cast<uint4*>(&sB[0][row * BPITCH + seg * 8]) = g;
        }
        __syncthreads();
        const unsigned short* sbuf = &sB[0][0];
#endif

        // A fragment: 16x32 f32 tile -> bf16 (ISA 16-bit A layout)
        union { unsigned int u[8]; v16bf v; } aF;
        const float* ap = aRow + k0 + khSel * 8;
        float4 f0 = *reinterpret_cast<const float4*>(ap);
        float4 f1 = *reinterpret_cast<const float4*>(ap + 4);
        float4 f2 = *reinterpret_cast<const float4*>(ap + 16);
        float4 f3 = *reinterpret_cast<const float4*>(ap + 20);
        aF.u[0] = pk_bf16(f0.x, f0.y); aF.u[1] = pk_bf16(f0.z, f0.w);
        aF.u[2] = pk_bf16(f1.x, f1.y); aF.u[3] = pk_bf16(f1.z, f1.w);
        aF.u[4] = pk_bf16(f2.x, f2.y); aF.u[5] = pk_bf16(f2.z, f2.w);
        aF.u[6] = pk_bf16(f3.x, f3.y); aF.u[7] = pk_bf16(f3.z, f3.w);

#pragma unroll
        for (int t = 0; t < 16; ++t) {
            union { uint4 q[2]; v16bf v; } bF;
            const unsigned short* bp = &sbuf[(t * 16 + r) * BPITCH + khSel * 8];
            bF.q[0] = *reinterpret_cast<const uint4*>(bp);
            bF.q[1] = *reinterpret_cast<const uint4*>(bp + 16);
            acc[t] = __builtin_amdgcn_wmma_f32_16x16x32_bf16(
                false, aF.v, false, bF.v, (short)0, acc[t], false, false);
        }
    }

    // store D: VGPR v holds M = v + 8*khSel, N = r
#pragma unroll
    for (int t = 0; t < 16; ++t) {
        float* op = out + (size_t)(rowBase + khSel * 8) * N + t * 16 + r;
#pragma unroll
        for (int v = 0; v < 8; ++v) op[(size_t)v * N] = acc[t][v];
    }
}

// ---------------- launch ----------------
extern "C" void kernel_launch(void* const* d_in, const int* in_sizes, int n_in,
                              void* d_out, int out_size, void* d_ws, size_t ws_size,
                              hipStream_t stream) {
    const float* inputs = (const float*)d_in[0];
    const float* PHIs   = (const float*)d_in[1];
    const float* THETAs = (const float*)d_in[2];
    const float* DCs    = (const float*)d_in[3];
    float* out = (float*)d_out;

    float* mzi = (float*)d_ws;                                   // 32640*8 f32
    size_t mziBytes = (size_t)NUM_UNITS * 8 * sizeof(float);     // multiple of 256
    unsigned short* powerBf = (unsigned short*)((char*)d_ws + mziBytes);  // 128 KB bf16

    mzi_kernel<<<(NUM_UNITS + 255) / 256, 256, 0, stream>>>(PHIs, THETAs, DCs, mzi);
    phasor_kernel<<<N / COLS, 256, 0, stream>>>(mzi, powerBf);
    gemm_kernel<<<BATCH / 128, 256, 0, stream>>>(inputs, powerBf, out);
}
